// CausalSelfAttention_18098992185701
// MI455X (gfx1250) — matrix operand order repaired and checked by reading, not compiled, not run on hardware
//
#include <hip/hip_runtime.h>
#include <hip/hip_bf16.h>
#include <math.h>
#include <stdint.h>

// Problem constants (match reference)
#define B_ 4
#define T_ 2048
#define C_ 1024
#define H_ 16
#define HD_ 64
#define ROWS_ (B_ * T_)          // 8192
#define NEG_BIG (-1.0e30f)

#define BK_ 32                   // K-step per WMMA
#define LDSB_STRIDE 40           // padded LDS row stride (elems): 80B -> bank-conflict-free b128 reads
#define LDSB_BUF (64 * LDSB_STRIDE)   // one B-tile buffer (64 cols x 32 k, padded)

typedef __attribute__((ext_vector_type(16))) __bf16 v16bf;
typedef __attribute__((ext_vector_type(8)))  float  v8f;

// ---------------------------------------------------------------------------
// WMMA fragment loaders (wave32, CDNA5 layouts from CDNA5 ISA)
// ---------------------------------------------------------------------------

// A-matrix 16x32 bf16: lane L -> row = L&15; half = L>>4.
// elems 0..7 : K = half*8 + e ; elems 8..15 : K = 16 + half*8 + (e-8)
__device__ __forceinline__ v16bf load_frag_a(const __bf16* base, int ld) {
  const int lane = threadIdx.x & 31;
  const __bf16* p = base + (lane & 15) * ld + (lane >> 4) * 8;
  v16bf a;
#pragma unroll
  for (int e = 0; e < 8; ++e) a[e] = p[e];
  p += 16;
#pragma unroll
  for (int e = 0; e < 8; ++e) a[8 + e] = p[e];
  return a;
}

// B-matrix 32x16 bf16, memory holds B^T ([N x K] row-major): elem (k,n) at
// base[n*ld + k]. lane L -> n = L&15 ; elems 0..7 : K = (L>>4)*8 + e ; elems 8..15 : K = 16 + (L>>4)*8 + (e-8).
__device__ __forceinline__ v16bf load_frag_bt(const __bf16* base, int ld) {
  const int lane = threadIdx.x & 31;
  const __bf16* p = base + (lane & 15) * ld + (lane >> 4) * 8;
  v16bf b;
#pragma unroll
  for (int e = 0; e < 8; ++e) b[e] = p[e];
#pragma unroll
  for (int e = 0; e < 8; ++e) b[8 + e] = p[16 + e];
  return b;
}

__device__ __forceinline__ v8f wmma_bf16(v16bf a, v16bf b, v8f c) {
  return __builtin_amdgcn_wmma_f32_16x16x32_bf16(false, a, false, b,
                                                 (short)0, c, false, false);
}

// ---------------------------------------------------------------------------
// CDNA5 async global->LDS copy (ASYNCcnt path) + wait
// ---------------------------------------------------------------------------
__device__ __forceinline__ void async_copy_b128(__bf16* lds_dst,
                                                const __bf16* gsrc) {
  // generic shared pointer: low 32 bits = LDS byte address
  uint32_t loff = (uint32_t)(uintptr_t)lds_dst;
  asm volatile("global_load_async_to_lds_b128 %0, %1, off"
               :: "v"(loff), "v"(gsrc)
               : "memory");
}

__device__ __forceinline__ void wait_async0() {
#if __has_builtin(__builtin_amdgcn_s_wait_asynccnt)
  __builtin_amdgcn_s_wait_asynccnt(0);
#else
  asm volatile("s_wait_asynccnt 0x0" ::: "memory");
#endif
}

// ---------------------------------------------------------------------------
// Shared GEMM body: one wave -> 16 rows x 64 cols; 8 waves/block share the
// same 64-col weight slab, staged in double-buffered LDS via async copies.
//   Arow : A base at (r0, 0), ld = C_   (per-wave)
//   Bw   : weights [N x K] row-major at (c0, 0), ld = C_ (per-block)
//   lds  : 2 * LDSB_BUF elements of __shared__
// ---------------------------------------------------------------------------
__device__ __forceinline__ void gemm_tile_16x64(const __bf16* __restrict__ Arow,
                                                const __bf16* __restrict__ Bw,
                                                __bf16* lds, v8f acc[4]) {
  const int tid   = threadIdx.x;        // 0..255
  const int col   = tid >> 2;           // 0..63
  const int chunk = tid & 3;            // 4 x 8-elem (16B) chunks per col row
  const __bf16* gsrc = Bw + col * C_ + chunk * 8;
  __bf16* ldst = lds + col * LDSB_STRIDE + chunk * 8;

  // prime buffer 0
  async_copy_b128(ldst, gsrc);
  wait_async0();
  __syncthreads();

  v16bf a = load_frag_a(Arow, C_);

  for (int k0 = 0; k0 < C_; k0 += BK_) {
    const int s = (k0 >> 5) & 1;
    const __bf16* cur = lds + s * LDSB_BUF;
    __bf16* nxt = lds + (s ^ 1) * LDSB_BUF;
    const bool more = (k0 + BK_) < C_;

    // overlap: fill next B tile (async DMA) + next A fragment (registers)
    if (more) {
      async_copy_b128(nxt + col * LDSB_STRIDE + chunk * 8, gsrc + k0 + BK_);
      __builtin_prefetch(Arow + k0 + 2 * BK_, 0, 3);   // near-scope prefetch
    }
    v16bf a_n;
    if (more) a_n = load_frag_a(Arow + k0 + BK_, C_);

    // consume current tile from LDS
    v16bf bf[4];
#pragma unroll
    for (int nt = 0; nt < 4; ++nt)
      bf[nt] = load_frag_bt(cur + nt * 16 * LDSB_STRIDE, LDSB_STRIDE);
#pragma unroll
    for (int nt = 0; nt < 4; ++nt)
      acc[nt] = wmma_bf16(a, bf[nt], acc[nt]);

    a = a_n;
    if (more) wait_async0();
    __syncthreads();
  }
}

// ---------------------------------------------------------------------------
// Kernel 0: f32 -> bf16 conversion
// ---------------------------------------------------------------------------
__global__ void cvt_bf16_kernel(const float* __restrict__ in,
                                __bf16* __restrict__ out, int n) {
  int i = blockIdx.x * blockDim.x + threadIdx.x;
  if (i < n) out[i] = (__bf16)in[i];
}

// ---------------------------------------------------------------------------
// Kernel 1: qkv = x @ w_qkv^T  (bf16 WMMA, f32 accum)
// Writes q (pre-scaled by 1/8) and k as [B,H,T,64]; v transposed [B,H,64,T].
// ---------------------------------------------------------------------------
__global__ void gemm_qkv_kernel(const __bf16* __restrict__ xb,
                                const __bf16* __restrict__ wqb,
                                __bf16* __restrict__ qb,
                                __bf16* __restrict__ kb,
                                __bf16* __restrict__ vtb) {
  __shared__ __bf16 ldsb[2 * LDSB_BUF];
  const int wave    = blockIdx.x * (blockDim.x >> 5) + (threadIdx.x >> 5);
  const int rowTile = wave % (ROWS_ / 16);
  const int colGrp  = wave / (ROWS_ / 16);
  const int r0 = rowTile * 16;
  const int c0 = colGrp * 64;

  v8f acc[4] = {v8f{}, v8f{}, v8f{}, v8f{}};
  gemm_tile_16x64(xb + r0 * C_, wqb + c0 * C_, ldsb, acc);

  const int lane = threadIdx.x & 31;
  const int n    = lane & 15;
  const int half = lane >> 4;
#pragma unroll
  for (int nt = 0; nt < 4; ++nt) {
#pragma unroll
    for (int v = 0; v < 8; ++v) {
      const int m = v + 8 * half;
      const int r = r0 + m;
      const int c = c0 + nt * 16 + n;
      const int b = r >> 11;        // / T_
      const int t = r & (T_ - 1);
      const float val = acc[nt][v];
      if (c < C_) {                                   // Q (pre-scaled 1/sqrt(64))
        const int h = c >> 6, d = c & 63;
        qb[((b * H_ + h) * T_ + t) * HD_ + d] = (__bf16)(val * 0.125f);
      } else if (c < 2 * C_) {                        // K
        const int cc = c - C_;
        const int h = cc >> 6, d = cc & 63;
        kb[((b * H_ + h) * T_ + t) * HD_ + d] = (__bf16)val;
      } else {                                        // V, transposed [d][t]
        const int cc = c - 2 * C_;
        const int h = cc >> 6, d = cc & 63;
        vtb[((b * H_ + h) * HD_ + d) * T_ + t] = (__bf16)val;
      }
    }
  }
}

// ---------------------------------------------------------------------------
// Kernel 2: causal flash attention, one wave per (b,h,16-row q tile).
// Scores via WMMA (q pre-scaled), online softmax in f32, P@V via WMMA with
// P re-laid C->A through a private 1KB LDS tile. Output y -> yb [B,T,C] bf16.
// ---------------------------------------------------------------------------
__global__ void attn_kernel(const __bf16* __restrict__ qb,
                            const __bf16* __restrict__ kb,
                            const __bf16* __restrict__ vtb,
                            __bf16* __restrict__ yb) {
  __shared__ __bf16 plds_all[8][16 * 32];
  __bf16* plds = plds_all[threadIdx.x >> 5];

  const int wave = blockIdx.x * (blockDim.x >> 5) + (threadIdx.x >> 5);
  const int qt = wave & (T_ / 16 - 1);   // 0..127
  const int bh = wave >> 7;              // 0..63
  const int b  = bh >> 4;
  const int h  = bh & 15;
  const int q0 = qt * 16;

  const __bf16* Q  = qb  + (size_t)bh * T_ * HD_;
  const __bf16* K  = kb  + (size_t)bh * T_ * HD_;
  const __bf16* VT = vtb + (size_t)bh * HD_ * T_;

  const int lane = threadIdx.x & 31;
  const int n    = lane & 15;
  const int half = lane >> 4;

  const v16bf qa0 = load_frag_a(Q + q0 * HD_,      HD_);
  const v16bf qa1 = load_frag_a(Q + q0 * HD_ + 32, HD_);

  float m_run[8], l_run[8];
#pragma unroll
  for (int v = 0; v < 8; ++v) { m_run[v] = NEG_BIG; l_run[v] = 0.0f; }
  v8f acc[4] = {v8f{}, v8f{}, v8f{}, v8f{}};

  const int kend = q0 + 16;              // keys 0 .. q0+15 participate
  for (int kblk = 0; kblk < kend; kblk += 32) {
    __builtin_prefetch(K + (kblk + 32) * HD_, 0, 3);
    // ---- scores: two 16x16 tiles over 32 keys, K-dim = HD = 64 (2 WMMAs ea)
    v8f s0 = {}, s1 = {};
    s0 = wmma_bf16(qa0, load_frag_bt(K + (kblk +  0) * HD_ +  0, HD_), s0);
    s0 = wmma_bf16(qa1, load_frag_bt(K + (kblk +  0) * HD_ + 32, HD_), s0);
    s1 = wmma_bf16(qa0, load_frag_bt(K + (kblk + 16) * HD_ +  0, HD_), s1);
    s1 = wmma_bf16(qa1, load_frag_bt(K + (kblk + 16) * HD_ + 32, HD_), s1);

    // ---- causal mask + online softmax (rows span 16 lanes)
#pragma unroll
    for (int v = 0; v < 8; ++v) {
      const int m  = v + 8 * half;
      const int qi = q0 + m;
      if (kblk + n      > qi) s0[v] = NEG_BIG;
      if (kblk + 16 + n > qi) s1[v] = NEG_BIG;

      float mx = fmaxf(s0[v], s1[v]);
      mx = fmaxf(mx, __shfl_xor(mx, 1, 16));
      mx = fmaxf(mx, __shfl_xor(mx, 2, 16));
      mx = fmaxf(mx, __shfl_xor(mx, 4, 16));
      mx = fmaxf(mx, __shfl_xor(mx, 8, 16));

      const float mn    = fmaxf(m_run[v], mx);
      const float alpha = __expf(m_run[v] - mn);
      const float p0    = __expf(s0[v] - mn);
      const float p1    = __expf(s1[v] - mn);

      float sm = p0 + p1;
      sm += __shfl_xor(sm, 1, 16);
      sm += __shfl_xor(sm, 2, 16);
      sm += __shfl_xor(sm, 4, 16);
      sm += __shfl_xor(sm, 8, 16);

      l_run[v] = l_run[v] * alpha + sm;
      m_run[v] = mn;
#pragma unroll
      for (int dt = 0; dt < 4; ++dt) acc[dt][v] *= alpha;

      plds[m * 32 + n]      = (__bf16)p0;   // P tile, C-layout -> LDS
      plds[m * 32 + 16 + n] = (__bf16)p1;
    }

    // ---- P(16x32) @ V(32x64): reload P in A layout, V^T is K-contiguous
    const v16bf pa = load_frag_a(plds, 32);
#pragma unroll
    for (int dt = 0; dt < 4; ++dt) {
      acc[dt] = wmma_bf16(pa, load_frag_bt(VT + (dt * 16) * T_ + kblk, T_),
                          acc[dt]);
    }
  }

  // ---- normalize and write y back in [B,T,C] bf16 for the proj GEMM
#pragma unroll
  for (int v = 0; v < 8; ++v) {
    const float inv_l = 1.0f / l_run[v];
    const int m = v + 8 * half;
    const int t = q0 + m;
#pragma unroll
    for (int dt = 0; dt < 4; ++dt) {
      const int c = h * HD_ + dt * 16 + n;
      yb[((size_t)(b * T_ + t)) * C_ + c] = (__bf16)(acc[dt][v] * inv_l);
    }
  }
}

// ---------------------------------------------------------------------------
// Kernel 3: out = y @ w_proj^T, f32 output. 512 row-tiles x 16 col-groups.
// ---------------------------------------------------------------------------
__global__ void gemm_proj_kernel(const __bf16* __restrict__ yb,
                                 const __bf16* __restrict__ wpb,
                                 float* __restrict__ out) {
  __shared__ __bf16 ldsb[2 * LDSB_BUF];
  const int wave    = blockIdx.x * (blockDim.x >> 5) + (threadIdx.x >> 5);
  const int rowTile = wave % (ROWS_ / 16);
  const int colGrp  = wave / (ROWS_ / 16);
  const int r0 = rowTile * 16;
  const int c0 = colGrp * 64;

  v8f acc[4] = {v8f{}, v8f{}, v8f{}, v8f{}};
  gemm_tile_16x64(yb + r0 * C_, wpb + c0 * C_, ldsb, acc);

  const int lane = threadIdx.x & 31;
  const int n    = lane & 15;
  const int half = lane >> 4;
#pragma unroll
  for (int nt = 0; nt < 4; ++nt) {
#pragma unroll
    for (int v = 0; v < 8; ++v) {
      const int m = v + 8 * half;
      out[(size_t)(r0 + m) * C_ + c0 + nt * 16 + n] = acc[nt][v];
    }
  }
}

// ---------------------------------------------------------------------------
// Launcher
// ---------------------------------------------------------------------------
extern "C" void kernel_launch(void* const* d_in, const int* in_sizes, int n_in,
                              void* d_out, int out_size, void* d_ws, size_t ws_size,
                              hipStream_t stream) {
  const float* x      = (const float*)d_in[0];   // [B,T,C]
  const float* w_qkv  = (const float*)d_in[1];   // [3C,C]
  const float* w_proj = (const float*)d_in[2];   // [C,C]
  float* out = (float*)d_out;                    // [B,T,C]

  char* ws = (char*)d_ws;
  const size_t XB  = (size_t)ROWS_ * C_;         // 8M elems
  const size_t WQ  = (size_t)3 * C_ * C_;        // 3M elems
  const size_t WP  = (size_t)C_ * C_;            // 1M elems
  const size_t QKV = (size_t)ROWS_ * C_;

  __bf16* xb  = (__bf16*)(ws);
  __bf16* wqb = (__bf16*)(ws + 2 * XB);
  __bf16* wpb = (__bf16*)(ws + 2 * (XB + WQ));
  __bf16* qb  = (__bf16*)(ws + 2 * (XB + WQ + WP));
  __bf16* kb  = (__bf16*)(ws + 2 * (XB + WQ + WP + QKV));
  __bf16* vtb = (__bf16*)(ws + 2 * (XB + WQ + WP + 2 * QKV));
  __bf16* yb  = (__bf16*)(ws + 2 * (XB + WQ + WP + 3 * QKV));

  // 0) f32 -> bf16 conversions
  cvt_bf16_kernel<<<(int)((XB + 255) / 256), 256, 0, stream>>>(x, xb, (int)XB);
  cvt_bf16_kernel<<<(int)((WQ + 255) / 256), 256, 0, stream>>>(w_qkv, wqb, (int)WQ);
  cvt_bf16_kernel<<<(int)((WP + 255) / 256), 256, 0, stream>>>(w_proj, wpb, (int)WP);

  // 1) QKV GEMM: 512 row-tiles * 48 col-groups = 24576 waves, 8 waves/block
  gemm_qkv_kernel<<<24576 / 8, 256, 0, stream>>>(xb, wqb, qb, kb, vtb);

  // 2) Flash attention: 64 heads * 128 q-tiles = 8192 waves
  attn_kernel<<<8192 / 8, 256, 0, stream>>>(qb, kb, vtb, yb);

  // 3) Projection GEMM: 512 * 16 = 8192 waves
  gemm_proj_kernel<<<8192 / 8, 256, 0, stream>>>(yb, wpb, out);
}
